// MemoryAugmented_87136296501511
// MI455X (gfx1250) — matrix operand order repaired and checked by reading, not compiled, not run on hardware
//
#include <hip/hip_runtime.h>

typedef __attribute__((ext_vector_type(16))) _Float16 v16h;
typedef __attribute__((ext_vector_type(8)))  _Float16 v8h;
typedef __attribute__((ext_vector_type(8)))  float    v8f;

#define B_ 64
#define L_ 12
#define N_ 883
#define D_ 64
#define T_ 12
// rows = B*N = 56512 = 3532 tiles of 16 = 883 blocks * 4 waves * 16 rows

// ---------- kernel 1: xs[b*N+n, d] = sum_l x[b,l,n,d]  (f32 -> f16) ----------
__global__ void reduce_l_kernel(const float* __restrict__ x, _Float16* __restrict__ xs) {
  int idx = blockIdx.x * blockDim.x + threadIdx.x;
  const int total = B_ * N_ * D_;
  if (idx >= total) return;
  int d = idx & 63;
  int n = (idx >> 6) % N_;
  int b = idx / (N_ * 64);
  size_t base = ((size_t)b * L_) * N_ * 64 + (size_t)n * 64 + d;
  float acc = 0.f;
#pragma unroll
  for (int l = 0; l < L_; ++l) acc += x[base + (size_t)l * N_ * 64];
  xs[idx] = (_Float16)acc;
}

// ---------- kernel 2: M -> f16 row-major and f16 transposed ----------
__global__ void mconv_kernel(const float* __restrict__ M,
                             _Float16* __restrict__ Mh, _Float16* __restrict__ MTh) {
  int idx = blockIdx.x * blockDim.x + threadIdx.x;
  if (idx >= T_ * 64 * 64) return;
  int d = idx & 63;
  int m = (idx >> 6) & 63;
  int t = idx >> 12;
  float v = M[idx];
  Mh[idx] = (_Float16)v;                      // [t][m][d]
  MTh[(t << 12) + (d << 6) + m] = (_Float16)v; // [t][d][m]
}

// ---------- helpers ----------
// A-matrix fragment (16x32 f16): lane = row (mod 16); half 0: K {k0..k0+7, k0+16..k0+23},
// half 1: K {k0+8..k0+15, k0+24..k0+31}. rowp points to a 64-element f16 row.
__device__ inline v16h load_a_frag(const _Float16* rowp, int k0, int half) {
  const v8h lo = *(const v8h*)(rowp + k0 + half * 8);
  const v8h hi = *(const v8h*)(rowp + k0 + 16 + half * 8);
  v16h r;
#pragma unroll
  for (int i = 0; i < 8; ++i) { r[i] = lo[i]; r[i + 8] = hi[i]; }
  return r;
}

// B-matrix fragment (32x16 f16): lane = column (mod 16); half 0: K {k0..k0+7, k0+16..k0+23},
// half 1: K {k0+8..k0+15, k0+24..k0+31}. colrow points to the 64-element row that is this column.
__device__ inline v16h load_b_frag(const _Float16* colrow, int k0, int half) {
  const v8h lo = *(const v8h*)(colrow + k0 + half * 8);
  const v8h hi = *(const v8h*)(colrow + k0 + 16 + half * 8);
  v16h r;
#pragma unroll
  for (int i = 0; i < 8; ++i) { r[i] = lo[i]; r[i + 8] = hi[i]; }
  return r;
}

template <int MASK>
__device__ inline float swz_xor(float v) {
  int x = __builtin_amdgcn_ds_swizzle(__builtin_bit_cast(int, v), (MASK << 10) | 0x1f);
  return __builtin_bit_cast(float, x);
}

// ---------- kernel 3: fused score-GEMM -> softmax -> value-GEMM ----------
__global__ __launch_bounds__(128) void mem_attn_kernel(
    const _Float16* __restrict__ xs,   // [56512][64]
    const _Float16* __restrict__ Mh,   // [T][64][64]  (m, d)
    const _Float16* __restrict__ MTh,  // [T][64][64]  (d, m)
    float* __restrict__ out) {         // [B][T][N][64]
  __shared__ __align__(32) _Float16 Msh[64 * 64];
  __shared__ __align__(32) _Float16 MTsh[64 * 64];
  __shared__ __align__(32) _Float16 scoreSh[4][16 * 64];

  const int t = blockIdx.y;
  const int tid = threadIdx.x;

  // stage M[t] and M[t]^T into LDS (8 KB each), vectorized
  {
    const uint4* g0 = (const uint4*)(Mh + (size_t)t * 4096);
    const uint4* g1 = (const uint4*)(MTh + (size_t)t * 4096);
    uint4* s0 = (uint4*)Msh;
    uint4* s1 = (uint4*)MTsh;
#pragma unroll
    for (int i = tid; i < 512; i += 128) { s0[i] = g0[i]; s1[i] = g1[i]; }
  }
  __syncthreads();

  const int w    = tid >> 5;
  const int lane = tid & 31;
  const int half = lane >> 4;
  const int lrow = lane & 15;
  const int tile = blockIdx.x * 4 + w;  // 16-row tile of the 56512 flattened (b,n) rows

  // ---- GEMM1: score(16x64) = xs_tile(16x64) x M[t]^T ----
  const _Float16* rowp = xs + (size_t)tile * 16 * 64 + (size_t)lrow * 64;
  v16h a0 = load_a_frag(rowp, 0, half);
  v16h a1 = load_a_frag(rowp, 32, half);

  v8f c[4] = {};  // score accumulators, m-tiles j=0..3 (cols j*16..j*16+15)
#pragma unroll
  for (int j = 0; j < 4; ++j) {
    const _Float16* mrow = Msh + (j * 16 + lrow) * 64;  // B column = row m of M[t]
    v16h b0 = load_b_frag(mrow, 0, half);
    v16h b1 = load_b_frag(mrow, 32, half);
    c[j] = __builtin_amdgcn_wmma_f32_16x16x32_f16(false, a0, false, b0, (short)0, c[j], false, false);
    c[j] = __builtin_amdgcn_wmma_f32_16x16x32_f16(false, a1, false, b1, (short)0, c[j], false, false);
  }

  // ---- softmax over m (64) per row, fully in registers ----
  // C-layout: VGPR v holds row (v + 8*half), columns striped over the 16 lanes of each half,
  // so xor-reduce with masks 1,2,4,8 stays inside a row.
#pragma unroll
  for (int v = 0; v < 8; ++v) {
    float mx = fmaxf(fmaxf(c[0][v], c[1][v]), fmaxf(c[2][v], c[3][v]));
    mx = fmaxf(mx, swz_xor<1>(mx));
    mx = fmaxf(mx, swz_xor<2>(mx));
    mx = fmaxf(mx, swz_xor<4>(mx));
    mx = fmaxf(mx, swz_xor<8>(mx));
    float e[4], s = 0.f;
#pragma unroll
    for (int j = 0; j < 4; ++j) {
      e[j] = __builtin_amdgcn_exp2f((c[j][v] - mx) * 1.44269504088896340736f);
      s += e[j];
    }
    s += swz_xor<1>(s);
    s += swz_xor<2>(s);
    s += swz_xor<4>(s);
    s += swz_xor<8>(s);
    float inv = __builtin_amdgcn_rcpf(s);
#pragma unroll
    for (int j = 0; j < 4; ++j) c[j][v] = e[j] * inv;
  }

  // ---- C-layout -> A-layout via LDS (f16 probabilities) ----
  _Float16* ss = scoreSh[w];
#pragma unroll
  for (int j = 0; j < 4; ++j)
#pragma unroll
    for (int v = 0; v < 8; ++v) {
      int row = v + 8 * half;
      ss[row * 64 + j * 16 + lrow] = (_Float16)c[j][v];
    }
  __syncthreads();

  // ---- GEMM2: value(16x64) = prob(16x64) x M[t] ----
  const _Float16* prow = ss + lrow * 64;
  v16h p0 = load_a_frag(prow, 0, half);
  v16h p1 = load_a_frag(prow, 32, half);

  v8f vc[4] = {};  // d-tiles jj=0..3
#pragma unroll
  for (int jj = 0; jj < 4; ++jj) {
    const _Float16* mtrow = MTsh + (jj * 16 + lrow) * 64;  // B column d = row d of M[t]^T
    v16h b0 = load_b_frag(mtrow, 0, half);
    v16h b1 = load_b_frag(mtrow, 32, half);
    vc[jj] = __builtin_amdgcn_wmma_f32_16x16x32_f16(false, p0, false, b0, (short)0, vc[jj], false, false);
    vc[jj] = __builtin_amdgcn_wmma_f32_16x16x32_f16(false, p1, false, b1, (short)0, vc[jj], false, false);
  }

  // ---- store value: out[b][t][n][d], f32; rows may straddle a b boundary (handled per element)
#pragma unroll
  for (int v = 0; v < 8; ++v) {
    int R = tile * 16 + v + 8 * half;  // flattened b*N + n
    int b = R / N_;
    int n = R - b * N_;
    float* op = out + (((size_t)b * T_ + t) * N_ + n) * 64;
#pragma unroll
    for (int jj = 0; jj < 4; ++jj) op[jj * 16 + lrow] = vc[jj][v];
  }
}

extern "C" void kernel_launch(void* const* d_in, const int* in_sizes, int n_in,
                              void* d_out, int out_size, void* d_ws, size_t ws_size,
                              hipStream_t stream) {
  const float* x = (const float*)d_in[0];  // [B][L][N][D] f32
  const float* M = (const float*)d_in[1];  // [T][64][64]  f32
  float* out = (float*)d_out;              // [B][T][N][64] f32

  char* ws = (char*)d_ws;
  _Float16* xs  = (_Float16*)ws;                                   // 56512*64 f16 = 7.23 MB
  _Float16* Mh  = (_Float16*)(ws + (size_t)56512 * 64 * 2);        // 12*4096 f16
  _Float16* MTh = (_Float16*)(ws + (size_t)56512 * 64 * 2 + (size_t)12 * 4096 * 2);

  const int total1 = B_ * N_ * D_;
  reduce_l_kernel<<<(total1 + 255) / 256, 256, 0, stream>>>(x, xs);
  mconv_kernel<<<(T_ * 4096 + 255) / 256, 256, 0, stream>>>(M, Mh, MTh);

  dim3 grid(N_ /*883 blocks*/, T_ /*12*/);
  mem_attn_kernel<<<grid, 128, 0, stream>>>(xs, Mh, MTh, out);
}
